// DFlashAttention_57380763074921
// MI455X (gfx1250) — compile-verified, matrix operand order checked by reading, not run on hardware
//
#include <hip/hip_runtime.h>
#include <hip/hip_bf16.h>

// ---------------------------------------------------------------------------
// Problem constants (from reference): B=1, S=2048, L=2048, T=4096,
// H=16, KVH=8 (GQA groups=2), HD=128, HID=2048.
// All GEMMs share K = 2048 (HID == H*HD), so it is a compile-time constant:
// weight-row addresses become immediate offsets (t * 64KiB) off one base.
// ---------------------------------------------------------------------------
#define S_ 2048
#define L_ 2048
#define T_ 4096
#define H_ 16
#define KVH_ 8
#define HD_ 128
#define HID_ 2048
#define KK_ 2048 /* contraction dim of every projection GEMM */
#define EPS_ 1e-6f
#define SCALE_ 0.08838834764831845f /* 1/sqrt(128) */

typedef _Float16 h16v __attribute__((ext_vector_type(16)));
typedef _Float16 h8v  __attribute__((ext_vector_type(8)));
typedef _Float16 h4v  __attribute__((ext_vector_type(4)));
typedef float    f8v  __attribute__((ext_vector_type(8)));

// WMMA: D = A(16x32 f16) x B(32x16 f16) + C(16x16 f32)
__device__ __forceinline__ f8v wmma_f16(h16v a, h16v b, f8v c) {
  return __builtin_amdgcn_wmma_f32_16x16x32_f16(false, a, false, b,
                                                (short)0, c, false, false);
}

// Build an A/B fragment (per-lane 16 halves) from two contiguous 8-half
// (16-byte) chunks -- matches the ISA 16-bit 16x32 operand layout:
// lane L: row = L&15, halves 0..7 = K[koff..koff+7], halves 8..15 = K[16+koff..].
__device__ __forceinline__ h16v ldfrag_h(const _Float16* p0, const _Float16* p1) {
  h8v a = *(const h8v*)p0;
  h8v b = *(const h8v*)p1;
  h16v f;
#pragma unroll
  for (int i = 0; i < 8; ++i) { f[i] = a[i]; f[8 + i] = b[i]; }
  return f;
}

// ---------------------------------------------------------------------------
// One-shot f32 -> f16 convert (8 elements per thread, b128 in / b128 out).
// Removes all cvt work from the GEMM hot loops and halves operand bytes.
// ---------------------------------------------------------------------------
__global__ __launch_bounds__(256)
void cvt_f32_f16_kernel(const float* __restrict__ src, _Float16* __restrict__ dst,
                        size_t n8) {
  const size_t i = (size_t)blockIdx.x * 256 + threadIdx.x;
  if (i >= n8) return;
  const float4 a = ((const float4*)src)[2 * i];
  const float4 b = ((const float4*)src)[2 * i + 1];
  h8v o;
  o[0] = (_Float16)a.x; o[1] = (_Float16)a.y; o[2] = (_Float16)a.z; o[3] = (_Float16)a.w;
  o[4] = (_Float16)b.x; o[5] = (_Float16)b.y; o[6] = (_Float16)b.z; o[7] = (_Float16)b.w;
  ((h8v*)dst)[i] = o;
}

// ---------------------------------------------------------------------------
// GEMM: C[M,N] = X[M,KK_] * W[N,KK_]^T, f16 operands, f32 WMMA accumulate,
// f32 output. One wave owns a 16x128 output tile (8 accumulators).
// All 8 B fragments of a k-step are loaded into distinct registers BEFORE
// the 8 WMMAs so the compiler can issue staged s_wait_loadcnt (overlap).
// ---------------------------------------------------------------------------
__global__ __launch_bounds__(256)
void gemm_h_xwt(const _Float16* __restrict__ X, const _Float16* __restrict__ W,
                float* __restrict__ C, int M, int N) {
  const int lane  = threadIdx.x & 31;
  const int wave  = threadIdx.x >> 5;
  const int gid   = blockIdx.x * 8 + wave;
  const int tilesM = M >> 4;
  const int mT = gid % tilesM;
  const int nB = gid / tilesM;
  const int m0 = mT * 16;
  const int n0 = nB * 128;
  if (n0 >= N) return;

  const int r    = lane & 15;
  const int kOff = (lane >> 4) * 8;

  f8v acc[8] = {};
  // Single base pointers: the 8 W rows differ by compile-time t*16*KK_ halves
  // (t * 64KiB), which fits the signed 24-bit instruction offset.
  const _Float16* xptr = X + (size_t)(m0 + r) * KK_ + kOff;
  const _Float16* wptr = W + (size_t)(n0 + r) * KK_ + kOff;

  for (int k = 0; k < KK_; k += 32) {
    __builtin_prefetch(xptr + k + 512, 0, 1); // global_prefetch_b8
    h16v a = ldfrag_h(xptr + k, xptr + k + 16);
    h16v b[8];
#pragma unroll
    for (int t = 0; t < 8; ++t)
      b[t] = ldfrag_h(wptr + t * 16 * KK_ + k, wptr + t * 16 * KK_ + k + 16);
#pragma unroll
    for (int t = 0; t < 8; ++t)
      acc[t] = wmma_f16(a, b[t], acc[t]);
  }

  // C/D layout: VGPR rr, lanes 0-15 -> [M=rr, N=lane]; lanes 16-31 -> [M=rr+8].
  const int hi = lane >> 4;
#pragma unroll
  for (int t = 0; t < 8; ++t)
#pragma unroll
    for (int rr = 0; rr < 8; ++rr)
      C[(size_t)(m0 + rr + 8 * hi) * N + n0 + t * 16 + r] = acc[t][rr];
}

// ---------------------------------------------------------------------------
// Fused RMS-norm + RoPE. One wave per (s, head) row of HD=128.
// dst[(h*dstT + dstRowOff + s)*HD + d], f16.
// ---------------------------------------------------------------------------
__global__ __launch_bounds__(256)
void normrope_kernel(const float* __restrict__ src, const float* __restrict__ w,
                     const long long* __restrict__ pos, _Float16* __restrict__ dst,
                     int nH, int dstT, int dstRowOff) {
  const int lane = threadIdx.x & 31;
  const int wave = threadIdx.x >> 5;
  const int gid  = blockIdx.x * 8 + wave;
  const int s  = gid / nH;
  const int hh = gid % nH;
  if (s >= S_) return;

  const float* x = src + ((size_t)s * nH + hh) * HD_;
  float4 xv = *(const float4*)(x + lane * 4);

  float ss = xv.x * xv.x + xv.y * xv.y + xv.z * xv.z + xv.w * xv.w;
  ss += __shfl_xor(ss, 1, 32);
  ss += __shfl_xor(ss, 2, 32);
  ss += __shfl_xor(ss, 4, 32);
  ss += __shfl_xor(ss, 8, 32);
  ss += __shfl_xor(ss, 16, 32);
  const float scale = rsqrtf(ss * (1.0f / HD_) + EPS_);

  const float p = (float)pos[s];
  float xn[4];
#pragma unroll
  for (int i = 0; i < 4; ++i) {
    const int d = lane * 4 + i;
    const float v = (i == 0 ? xv.x : i == 1 ? xv.y : i == 2 ? xv.z : xv.w);
    xn[i] = v * scale * w[d];
  }
  h4v out;
#pragma unroll
  for (int i = 0; i < 4; ++i) {
    const int d = lane * 4 + i;
    const float partner = __shfl_xor(xn[i], 16, 32);    // value at d ^ 64
    const float rot = (lane < 16) ? -partner : partner; // rotate_half
    const int j = d & 63;
    const float freq = p * __powf(10000.0f, -(float)j * (1.0f / 64.0f));
    out[i] = (_Float16)(xn[i] * __cosf(freq) + rot * __sinf(freq));
  }
  _Float16* drow = dst + ((size_t)hh * dstT + dstRowOff + s) * HD_;
  *(h4v*)(drow + lane * 4) = out;
}

// K_ctx f32 [L, KVH*HD] -> Kf f16 [KVH][T][HD] rows 0..L-1 (no norm/rope).
__global__ __launch_bounds__(256)
void convert_kctx_kernel(const float* __restrict__ src, _Float16* __restrict__ Kf) {
  const size_t i = (size_t)blockIdx.x * 256 + threadIdx.x;
  const int d   = (int)(i % HD_);
  const int kvh = (int)((i / HD_) % KVH_);
  const int l   = (int)(i / (HD_ * KVH_));
  Kf[((size_t)kvh * T_ + l) * HD_ + d] = (_Float16)src[i];
}

// V_ctx / V f32 [*, KVH*HD] -> transposed Vt f16 [KVH][HD][T]
// (pre-transposed so the P*V B-operand is per-lane contiguous).
__global__ __launch_bounds__(256)
void convert_v_kernel(const float* __restrict__ Vctx, const float* __restrict__ Vself,
                      _Float16* __restrict__ Vt) {
  const size_t i = (size_t)blockIdx.x * 256 + threadIdx.x;
  const int t   = (int)(i % T_);
  const int d   = (int)((i / T_) % HD_);
  const int kvh = (int)(i / ((size_t)T_ * HD_));
  const float v = (t < L_)
      ? Vctx[((size_t)t * KVH_ + kvh) * HD_ + d]
      : Vself[((size_t)(t - L_) * KVH_ + kvh) * HD_ + d];
  Vt[i] = (_Float16)v;
}

// ---------------------------------------------------------------------------
// Flash attention: one wave per (head, 16-query tile). Online softmax,
// f16 WMMA for Q*K^T and P*V, f32 accumulation. P goes through a per-wave
// LDS tile to convert the C/D layout into the A-operand layout.
// Output written as f16 so the output projection is a pure f16 GEMM.
// ---------------------------------------------------------------------------
__global__ __launch_bounds__(256)
void attn_kernel(const _Float16* __restrict__ Qp, const _Float16* __restrict__ Kf,
                 const _Float16* __restrict__ Vt, _Float16* __restrict__ Oh) {
  __shared__ __align__(16) _Float16 Pl[8][16][72]; // per-wave 16x64 P tile (padded)

  const int lane = threadIdx.x & 31;
  const int wave = threadIdx.x >> 5;
  const int gid  = blockIdx.x * 8 + wave;      // 0 .. H*(S/16)-1
  const int h    = gid >> 7;                   // 128 query tiles per head
  const int qT   = gid & 127;
  const int kvh  = h >> 1;                     // GQA: 2 q-heads per kv-head
  const int q0   = qT * 16;

  const int r    = lane & 15;
  const int hi   = lane >> 4;
  const int kOff = hi * 8;

  // Q fragments: 4 k-chunks of 32 over HD=128.
  h16v qf[4];
  const _Float16* qrow = Qp + ((size_t)h * S_ + q0 + r) * HD_ + kOff;
#pragma unroll
  for (int s = 0; s < 4; ++s)
    qf[s] = ldfrag_h(qrow + 32 * s, qrow + 32 * s + 16);

  float mrow[8], lrow[8];
  f8v acc[8] = {};
#pragma unroll
  for (int rr = 0; rr < 8; ++rr) { mrow[rr] = -1e30f; lrow[rr] = 0.0f; }

  const _Float16* Kb = Kf + (size_t)kvh * T_ * HD_ + (size_t)r * HD_ + kOff;
  const _Float16* Vb = Vt + (size_t)kvh * HD_ * T_ + (size_t)r * T_ + kOff;

  for (int t0 = 0; t0 < T_; t0 += 64) {
    // ---- scores: 4 tiles of 16 keys, contraction over HD ----
    // kptr advances once per block; the nt/s offsets are immediates.
    const _Float16* kptr = Kb + (size_t)t0 * HD_;
    f8v sc[4] = {};
#pragma unroll
    for (int nt = 0; nt < 4; ++nt) {
      h16v bf[4];
#pragma unroll
      for (int s = 0; s < 4; ++s)
        bf[s] = ldfrag_h(kptr + nt * 16 * HD_ + 32 * s,
                         kptr + nt * 16 * HD_ + 32 * s + 16);
#pragma unroll
      for (int s = 0; s < 4; ++s) sc[nt] = wmma_f16(qf[s], bf[s], sc[nt]);
    }
#pragma unroll
    for (int nt = 0; nt < 4; ++nt)
#pragma unroll
      for (int rr = 0; rr < 8; ++rr) sc[nt][rr] *= SCALE_;

    // ---- row max over the 64-key block (reduce across 16-lane C/D group) ----
    float pscale[8];
#pragma unroll
    for (int rr = 0; rr < 8; ++rr) {
      float v = fmaxf(fmaxf(sc[0][rr], sc[1][rr]), fmaxf(sc[2][rr], sc[3][rr]));
      v = fmaxf(v, __shfl_xor(v, 1, 32));
      v = fmaxf(v, __shfl_xor(v, 2, 32));
      v = fmaxf(v, __shfl_xor(v, 4, 32));
      v = fmaxf(v, __shfl_xor(v, 8, 32));
      const float mnew = fmaxf(mrow[rr], v);
      pscale[rr] = __expf(mrow[rr] - mnew);
      mrow[rr] = mnew;
      lrow[rr] *= pscale[rr];
    }
#pragma unroll
    for (int nt = 0; nt < 8; ++nt)
#pragma unroll
      for (int rr = 0; rr < 8; ++rr) acc[nt][rr] *= pscale[rr];

    // ---- P = exp(s - m): stash f16 tile in LDS, accumulate row sums ----
#pragma unroll
    for (int nt = 0; nt < 4; ++nt) {
      float rs[8];
#pragma unroll
      for (int rr = 0; rr < 8; ++rr) {
        const float pv = __expf(sc[nt][rr] - mrow[rr]);
        rs[rr] = pv;
        Pl[wave][rr + 8 * hi][nt * 16 + r] = (_Float16)pv;
      }
#pragma unroll
      for (int rr = 0; rr < 8; ++rr) {
        float v = rs[rr];
        v += __shfl_xor(v, 1, 32);
        v += __shfl_xor(v, 2, 32);
        v += __shfl_xor(v, 4, 32);
        v += __shfl_xor(v, 8, 32);
        lrow[rr] += v;
      }
    }
    asm volatile("s_wait_dscnt 0x0" ::: "memory"); // wave-local LDS RAW fence

    // ---- reload P in A-operand layout, then P*V (Vt is pre-transposed) ----
    h16v pf[2];
#pragma unroll
    for (int c = 0; c < 2; ++c)
      pf[c] = ldfrag_h(&Pl[wave][r][c * 32 + kOff], &Pl[wave][r][c * 32 + 16 + kOff]);
    const _Float16* vptr = Vb + t0;
#pragma unroll
    for (int nt = 0; nt < 8; ++nt) {
      h16v bf[2];
#pragma unroll
      for (int c = 0; c < 2; ++c)
        bf[c] = ldfrag_h(vptr + nt * 16 * T_ + c * 32,
                         vptr + nt * 16 * T_ + c * 32 + 16);
#pragma unroll
      for (int c = 0; c < 2; ++c) acc[nt] = wmma_f16(pf[c], bf[c], acc[nt]);
    }
  }

  // ---- epilogue: normalize and write Oh f16 [S][H*HD] ----
#pragma unroll
  for (int nt = 0; nt < 8; ++nt)
#pragma unroll
    for (int rr = 0; rr < 8; ++rr) {
      const float o = acc[nt][rr] / lrow[rr];
      Oh[(size_t)(q0 + rr + 8 * hi) * (H_ * HD_) + h * HD_ + nt * 16 + r] =
          (_Float16)o;
    }
}

// ---------------------------------------------------------------------------
// Host-side launch
// ---------------------------------------------------------------------------
extern "C" void kernel_launch(void* const* d_in, const int* in_sizes, int n_in,
                              void* d_out, int out_size, void* d_ws, size_t ws_size,
                              hipStream_t stream) {
  (void)in_sizes; (void)n_in; (void)out_size; (void)ws_size;

  const float*     hidden = (const float*)d_in[0];
  const float*     ctx    = (const float*)d_in[1];
  const long long* pos    = (const long long*)d_in[2];
  const float*     Wq     = (const float*)d_in[3];
  const float*     Wk     = (const float*)d_in[4];
  const float*     Wv     = (const float*)d_in[5];
  const float*     Wo     = (const float*)d_in[6];
  const float*     Wkc    = (const float*)d_in[7];
  const float*     Wvc    = (const float*)d_in[8];
  const float*     qw     = (const float*)d_in[9];
  const float*     kw     = (const float*)d_in[10];
  float* out = (float*)d_out;

  // Workspace layout
  char* ws = (char*)d_ws;
  size_t off = 0;
  // f16 copies of activations + weights (one-shot convert)
  _Float16* Xh  = (_Float16*)(ws + off); off += (size_t)S_ * HID_ * 2;          // 8 MiB
  _Float16* Ch  = (_Float16*)(ws + off); off += (size_t)L_ * HID_ * 2;          // 8 MiB
  _Float16* Wqh = (_Float16*)(ws + off); off += (size_t)H_ * HD_ * HID_ * 2;    // 8 MiB
  _Float16* Wkh = (_Float16*)(ws + off); off += (size_t)KVH_ * HD_ * HID_ * 2;  // 4 MiB
  _Float16* Wvh = (_Float16*)(ws + off); off += (size_t)KVH_ * HD_ * HID_ * 2;  // 4 MiB
  _Float16* Woh = (_Float16*)(ws + off); off += (size_t)HID_ * H_ * HD_ * 2;    // 8 MiB
  _Float16* Wkch = (_Float16*)(ws + off); off += (size_t)KVH_ * HD_ * HID_ * 2; // 4 MiB
  _Float16* Wvch = (_Float16*)(ws + off); off += (size_t)KVH_ * HD_ * HID_ * 2; // 4 MiB
  // f32 projection outputs (pre norm/rope)
  float* Qraw  = (float*)(ws + off); off += (size_t)S_ * H_ * HD_ * 4;          // 16 MiB
  float* Kraw  = (float*)(ws + off); off += (size_t)S_ * KVH_ * HD_ * 4;        //  8 MiB
  float* Vraw  = (float*)(ws + off); off += (size_t)S_ * KVH_ * HD_ * 4;        //  8 MiB
  float* Kcraw = (float*)(ws + off); off += (size_t)L_ * KVH_ * HD_ * 4;        //  8 MiB
  float* Vcraw = (float*)(ws + off); off += (size_t)L_ * KVH_ * HD_ * 4;        //  8 MiB
  // f16 attention operands / output
  _Float16* Qp = (_Float16*)(ws + off); off += (size_t)H_ * S_ * HD_ * 2;       //  8 MiB
  _Float16* Kf = (_Float16*)(ws + off); off += (size_t)KVH_ * T_ * HD_ * 2;     //  8 MiB
  _Float16* Vt = (_Float16*)(ws + off); off += (size_t)KVH_ * HD_ * T_ * 2;     //  8 MiB
  _Float16* Oh = (_Float16*)(ws + off); off += (size_t)S_ * H_ * HD_ * 2;       //  8 MiB

  // 0) One-shot f32 -> f16 converts (b128 in / b128 out, fully coalesced)
  const size_t big8   = (size_t)S_ * HID_ / 8;         // 512K threads
  const size_t small8 = (size_t)KVH_ * HD_ * HID_ / 8; // 256K threads
  cvt_f32_f16_kernel<<<big8 / 256, 256, 0, stream>>>(hidden, Xh, big8);
  cvt_f32_f16_kernel<<<big8 / 256, 256, 0, stream>>>(ctx, Ch, big8);
  cvt_f32_f16_kernel<<<big8 / 256, 256, 0, stream>>>(Wq, Wqh, big8);
  cvt_f32_f16_kernel<<<small8 / 256, 256, 0, stream>>>(Wk, Wkh, small8);
  cvt_f32_f16_kernel<<<small8 / 256, 256, 0, stream>>>(Wv, Wvh, small8);
  cvt_f32_f16_kernel<<<big8 / 256, 256, 0, stream>>>(Wo, Woh, big8);
  cvt_f32_f16_kernel<<<small8 / 256, 256, 0, stream>>>(Wkc, Wkch, small8);
  cvt_f32_f16_kernel<<<small8 / 256, 256, 0, stream>>>(Wvc, Wvch, small8);

  // 1) Projections (C = X * W^T), pure f16-operand WMMA GEMMs
  // grid = (M/16) * (N/128) / 8 waves-per-block
  gemm_h_xwt<<<(S_ / 16) * (HID_ / 128) / 8, 256, 0, stream>>>(Xh, Wqh, Qraw, S_, H_ * HD_);
  gemm_h_xwt<<<(S_ / 16) * (KVH_ * HD_ / 128) / 8, 256, 0, stream>>>(Xh, Wkh, Kraw, S_, KVH_ * HD_);
  gemm_h_xwt<<<(S_ / 16) * (KVH_ * HD_ / 128) / 8, 256, 0, stream>>>(Xh, Wvh, Vraw, S_, KVH_ * HD_);
  gemm_h_xwt<<<(L_ / 16) * (KVH_ * HD_ / 128) / 8, 256, 0, stream>>>(Ch, Wkch, Kcraw, L_, KVH_ * HD_);
  gemm_h_xwt<<<(L_ / 16) * (KVH_ * HD_ / 128) / 8, 256, 0, stream>>>(Ch, Wvch, Vcraw, L_, KVH_ * HD_);

  // 2) RMS-norm + RoPE -> head-major f16
  normrope_kernel<<<(S_ * H_) / 8, 256, 0, stream>>>(Qraw, qw, pos, Qp, H_, S_, 0);
  normrope_kernel<<<(S_ * KVH_) / 8, 256, 0, stream>>>(Kraw, kw, pos, Kf, KVH_, T_, L_);

  // 3) Context K (plain convert) and V (transposed convert)
  convert_kctx_kernel<<<((size_t)L_ * KVH_ * HD_) / 256, 256, 0, stream>>>(Kcraw, Kf);
  convert_v_kernel<<<((size_t)KVH_ * HD_ * T_) / 256, 256, 0, stream>>>(Vcraw, Vraw, Vt);

  // 4) Flash attention -> Oh f16 [S][H*HD]
  attn_kernel<<<(H_ * (S_ / 16)) / 8, 256, 0, stream>>>(Qp, Kf, Vt, Oh);

  // 5) Output projection -> d_out (f16 A from attention, f16 W)
  gemm_h_xwt<<<(S_ / 16) * (HID_ / 128) / 8, 256, 0, stream>>>(Oh, Woh, out, S_, HID_);
}
